// EFLayoutSpatialAttention_36661840838668
// MI455X (gfx1250) — matrix operand order repaired and checked by reading, not compiled, not run on hardware
//
#include <hip/hip_runtime.h>

#define SEQ 4096
#define DIM 1024
#define NH  16
#define HD  64

typedef __attribute__((ext_vector_type(16))) _Float16     v16h;
typedef __attribute__((ext_vector_type(8)))  float        v8f;
typedef __attribute__((ext_vector_type(4)))  unsigned int v4u;
typedef __attribute__((ext_vector_type(4)))  int          i4;

#if __has_builtin(__builtin_amdgcn_global_load_async_to_lds_b128) && \
    __has_builtin(__builtin_amdgcn_s_wait_asynccnt)
#define ASYNC_LDS 1
typedef __attribute__((address_space(1))) i4* gp4;   // global int4*
typedef __attribute__((address_space(3))) i4* sp4;   // LDS int4*
#else
#define ASYNC_LDS 0
#endif

#if __has_builtin(__builtin_amdgcn_sched_barrier)
#define SCHED_FENCE() __builtin_amdgcn_sched_barrier(0)
#else
#define SCHED_FENCE()
#endif

union FragH {
    v16h h;
    struct { v4u lo; v4u hi; } u;
    unsigned int w[8];
    _Float16 e[16];
};

__device__ __forceinline__ v8f zero8() {
    v8f z;
    for (int i = 0; i < 8; ++i) z[i] = 0.0f;
    return z;
}

__device__ __forceinline__ v8f wmma_f16(const FragH& a, const FragH& b, v8f c) {
    return __builtin_amdgcn_wmma_f32_16x16x32_f16(
        false, a.h, false, b.h, (short)0, c, false, false);
}

// ---------------------------------------------------------------------------
// Kernel 1: QKV = H @ Wqkv^T + b, RoPE on Q/K (Q pre-scaled), f16 outputs
// Qh[h][seq][d], Kh[h][seq][d], Vt[h][d][seq].
// 256 threads (8 waves), tile 128x128, K-step 32, reg-staged double buffer
// with sched fences pinning: gload(next) | frag loads | wmma | lstore.
// ---------------------------------------------------------------------------
__global__ void __launch_bounds__(256) qkv_rope_kernel(
    const float* __restrict__ H, const float* __restrict__ cosp,
    const float* __restrict__ sinp, const float* __restrict__ W,
    const float* __restrict__ bias,
    _Float16* __restrict__ Qh, _Float16* __restrict__ Kh,
    _Float16* __restrict__ Vt)
{
    __shared__ _Float16 As[2][128][40];   // [m][k], 80B rows (16B aligned)
    __shared__ _Float16 Bs[2][128][40];   // [n][k]

    const int tid  = threadIdx.x;
    const int wave = tid >> 5;
    const int lane = tid & 31;
    const int l16  = lane & 15;
    const int h8   = lane >> 4;
    const int m0   = blockIdx.y * 128;
    const int n0   = blockIdx.x * 128;

    v8f acc[8];
    for (int t = 0; t < 8; ++t) acc[t] = zero8();

    float4 ra[4], rb[4];
    auto gload = [&](int k0) {
        for (int ii = 0; ii < 4; ++ii) {
            const int i = tid + ii * 256;
            const int r = i >> 3, c4 = (i & 7) << 2;
            ra[ii] = *reinterpret_cast<const float4*>(
                H + (size_t)(m0 + r) * DIM + k0 + c4);
            rb[ii] = *reinterpret_cast<const float4*>(
                W + (size_t)(n0 + r) * DIM + k0 + c4);
        }
    };
    auto lstore = [&](int buf) {
        for (int ii = 0; ii < 4; ++ii) {
            const int i = tid + ii * 256;
            const int r = i >> 3, c4 = (i & 7) << 2;
            As[buf][r][c4 + 0] = (_Float16)ra[ii].x;
            As[buf][r][c4 + 1] = (_Float16)ra[ii].y;
            As[buf][r][c4 + 2] = (_Float16)ra[ii].z;
            As[buf][r][c4 + 3] = (_Float16)ra[ii].w;
            Bs[buf][r][c4 + 0] = (_Float16)rb[ii].x;
            Bs[buf][r][c4 + 1] = (_Float16)rb[ii].y;
            Bs[buf][r][c4 + 2] = (_Float16)rb[ii].z;
            Bs[buf][r][c4 + 3] = (_Float16)rb[ii].w;
        }
    };
    auto compute = [&](int cur) {
        FragH a;
        const int m = (wave << 4) + l16;
        a.u.lo = *reinterpret_cast<const v4u*>(&As[cur][m][(h8 << 3)]);
        a.u.hi = *reinterpret_cast<const v4u*>(&As[cur][m][16 + (h8 << 3)]);
        FragH bf[8];
        for (int t = 0; t < 8; ++t) {
            const int n = (t << 4) + l16;
            bf[t].u.lo = *reinterpret_cast<const v4u*>(&Bs[cur][n][(h8 << 3)]);
            bf[t].u.hi = *reinterpret_cast<const v4u*>(&Bs[cur][n][16 + (h8 << 3)]);
        }
        SCHED_FENCE();   // ds loads issue as a clause, single wait below
        for (int t = 0; t < 8; ++t) acc[t] = wmma_f16(a, bf[t], acc[t]);
    };

    gload(0);
    lstore(0);
    __syncthreads();

    for (int kt = 0; kt < DIM / 32 - 1; ++kt) {
        const int cur = kt & 1;
        gload(32 * (kt + 1));      // global loads in flight during wmma
        SCHED_FENCE();
        compute(cur);
        SCHED_FENCE();
        lstore(1 - cur);           // waits loads here, after compute
        __syncthreads();
    }
    compute((DIM / 32 - 1) & 1);   // last K-step, no staging

    // epilogue: bias, RoPE, scatter (partner d+/-32 lives in tile t^2, same lane)
    float val[8][8];
    for (int t = 0; t < 8; ++t) {
        const float bt = bias[n0 + (t << 4) + l16];
        for (int r = 0; r < 8; ++r) val[t][r] = acc[t][r] + bt;
    }

    const int sec   = n0 / DIM;   // 0=Q 1=K 2=V (uniform per block)
    const int cbase = n0 % DIM;
    for (int t = 0; t < 8; ++t) {
        const int c  = cbase + (t << 4) + l16;
        const int hh = c >> 6;
        const int d  = c & 63;
        for (int r = 0; r < 8; ++r) {
            const int grow = m0 + (wave << 4) + r + (h8 << 3);
            const float v = val[t][r];
            if (sec == 2) {
                Vt[(size_t)hh * HD * SEQ + (size_t)d * SEQ + grow] = (_Float16)v;
            } else {
                const float ct = cosp[grow * HD + d];
                const float st = sinp[grow * HD + d];
                const float part = val[t ^ 2][r];
                const float rot  = (d < 32) ? -part : part;
                float o = v * ct + rot * st;
                if (sec == 0) {
                    o *= 0.125f;
                    Qh[(size_t)hh * SEQ * HD + (size_t)grow * HD + d] = (_Float16)o;
                } else {
                    Kh[(size_t)hh * SEQ * HD + (size_t)grow * HD + d] = (_Float16)o;
                }
            }
        }
    }
}

// ---------------------------------------------------------------------------
// Kernel 2: flash attention. Block = (128 q-rows, one head), 8 waves x 16 rows.
// K/Vt tiles staged in LDS per block (async double-buffered when available).
// ---------------------------------------------------------------------------
__global__ void __launch_bounds__(256) attn_kernel(
    const _Float16* __restrict__ Qh, const _Float16* __restrict__ Kh,
    const _Float16* __restrict__ Vt, _Float16* __restrict__ AO)
{
    __shared__ _Float16 Ks[2][64][72];   // [key][d], 144B rows
    __shared__ _Float16 Vs[2][64][72];   // [d][key]
    __shared__ _Float16 Ps[8][16][72];   // per-wave P tile

    const int tid  = threadIdx.x;
    const int wave = tid >> 5;
    const int lane = tid & 31;
    const int l16  = lane & 15;
    const int h8   = lane >> 4;
    const int h    = blockIdx.y;
    const int q0   = blockIdx.x * 128;
    const int mrow = q0 + (wave << 4) + l16;

    const _Float16* Kbase = Kh + (size_t)h * SEQ * HD;
    const _Float16* Vbase = Vt + (size_t)h * HD * SEQ;

    // Q fragments, live across the kv loop
    const _Float16* Qrow = Qh + (size_t)h * SEQ * HD + (size_t)mrow * HD;
    FragH aq[2];
    for (int kc = 0; kc < 2; ++kc) {
        aq[kc].u.lo = *reinterpret_cast<const v4u*>(Qrow + kc * 32 + (h8 << 3));
        aq[kc].u.hi = *reinterpret_cast<const v4u*>(Qrow + kc * 32 + 16 + (h8 << 3));
    }

    float rowmax[8], rowsum[8];
    for (int r = 0; r < 8; ++r) { rowmax[r] = -3.0e38f; rowsum[r] = 0.0f; }
    v8f oacc[4];
    for (int t = 0; t < 4; ++t) oacc[t] = zero8();

#if ASYNC_LDS
    auto issue_tile = [&](int it, int buf) {
        const int kv = it * 64;
        for (int ii = 0; ii < 2; ++ii) {
            const int i = tid + ii * 256;           // 0..511: 64 rows x 8 x16B
            const int r = i >> 3, c8 = (i & 7) << 3;
            __builtin_amdgcn_global_load_async_to_lds_b128(
                (gp4)(Kbase + (size_t)(kv + r) * HD + c8),
                (sp4)(&Ks[buf][r][c8]), 0, 0);
            __builtin_amdgcn_global_load_async_to_lds_b128(
                (gp4)(Vbase + (size_t)r * SEQ + kv + c8),
                (sp4)(&Vs[buf][r][c8]), 0, 0);
        }
    };
    issue_tile(0, 0);
#else
    auto stage_tile = [&](int kv) {
        for (int ii = 0; ii < 2; ++ii) {
            const int i = tid + ii * 256;
            const int r = i >> 3, c8 = (i & 7) << 3;
            *reinterpret_cast<v4u*>(&Ks[0][r][c8]) =
                *reinterpret_cast<const v4u*>(Kbase + (size_t)(kv + r) * HD + c8);
            *reinterpret_cast<v4u*>(&Vs[0][r][c8]) =
                *reinterpret_cast<const v4u*>(Vbase + (size_t)r * SEQ + kv + c8);
        }
    };
#endif

    for (int it = 0; it < SEQ / 64; ++it) {
#if ASYNC_LDS
        const int cur = it & 1;
        __builtin_amdgcn_s_wait_asynccnt(0);   // my copies of tile `it` done
        __syncthreads();                       // all threads' copies visible
        if (it + 1 < SEQ / 64) issue_tile(it + 1, 1 - cur);  // overlap next
        SCHED_FENCE();
#else
        const int cur = 0;
        __syncthreads();
        stage_tile(it * 64);
        __syncthreads();
#endif

        // ---- S = Q * K^T : K fragment loads as a clause, 8 back-to-back WMMA
        FragH bk[8];
        for (int nt = 0; nt < 4; ++nt) {
            const int key = (nt << 4) + l16;
            for (int kc = 0; kc < 2; ++kc) {
                bk[nt * 2 + kc].u.lo = *reinterpret_cast<const v4u*>(
                    &Ks[cur][key][kc * 32 + (h8 << 3)]);
                bk[nt * 2 + kc].u.hi = *reinterpret_cast<const v4u*>(
                    &Ks[cur][key][kc * 32 + 16 + (h8 << 3)]);
            }
        }
        SCHED_FENCE();
        v8f s[4];
        for (int t = 0; t < 4; ++t) s[t] = zero8();
        for (int nt = 0; nt < 4; ++nt)
            for (int kc = 0; kc < 2; ++kc)
                s[nt] = wmma_f16(aq[kc], bk[nt * 2 + kc], s[nt]);
        SCHED_FENCE();

        // ---- V fragments loaded early: overlap with softmax VALU chains
        FragH bv[8];
        for (int dt = 0; dt < 4; ++dt) {
            const int d = (dt << 4) + l16;
            for (int kc = 0; kc < 2; ++kc) {
                bv[dt * 2 + kc].u.lo = *reinterpret_cast<const v4u*>(
                    &Vs[cur][d][kc * 32 + (h8 << 3)]);
                bv[dt * 2 + kc].u.hi = *reinterpret_cast<const v4u*>(
                    &Vs[cur][d][kc * 32 + 16 + (h8 << 3)]);
            }
        }
        SCHED_FENCE();

        // ---- online softmax (lane-half owns rows r + 8*h8)
        float fac[8];
        for (int r = 0; r < 8; ++r) {
            float mx = fmaxf(fmaxf(s[0][r], s[1][r]), fmaxf(s[2][r], s[3][r]));
            for (int off = 1; off < 16; off <<= 1)
                mx = fmaxf(mx, __shfl_xor(mx, off, 32));
            const float nm = fmaxf(rowmax[r], mx);
            fac[r] = __expf(rowmax[r] - nm);
            rowmax[r] = nm;
        }
        for (int r = 0; r < 8; ++r) {
            const int m = r + (h8 << 3);
            float ps = 0.0f;
            for (int nt = 0; nt < 4; ++nt) {
                const float p = __expf(s[nt][r] - rowmax[r]);
                ps += p;
                Ps[wave][m][(nt << 4) + l16] = (_Float16)p;
            }
            for (int off = 1; off < 16; off <<= 1)
                ps += __shfl_xor(ps, off, 32);
            rowsum[r] = rowsum[r] * fac[r] + ps;
        }
        for (int t = 0; t < 4; ++t)
            for (int r = 0; r < 8; ++r)
                oacc[t][r] *= fac[r];

        // ---- O += P * V (P read back in A layout; DS in-order within wave)
        for (int kc = 0; kc < 2; ++kc) {
            FragH ap;
            ap.u.lo = *reinterpret_cast<const v4u*>(
                &Ps[wave][l16][kc * 32 + (h8 << 3)]);
            ap.u.hi = *reinterpret_cast<const v4u*>(
                &Ps[wave][l16][kc * 32 + 16 + (h8 << 3)]);
            for (int dt = 0; dt < 4; ++dt)
                oacc[dt] = wmma_f16(ap, bv[dt * 2 + kc], oacc[dt]);
        }
    }

    for (int dt = 0; dt < 4; ++dt)
        for (int r = 0; r < 8; ++r) {
            const float o = oacc[dt][r] / rowsum[r];
            const int m = q0 + (wave << 4) + r + (h8 << 3);
            AO[(size_t)m * DIM + h * HD + (dt << 4) + l16] = (_Float16)o;
        }
}

// ---------------------------------------------------------------------------
// Kernel 3: out = AO @ proj_w^T + proj_b (f16 A, f32->f16 W, f32 out)
// ---------------------------------------------------------------------------
__global__ void __launch_bounds__(256) proj_kernel(
    const _Float16* __restrict__ AO, const float* __restrict__ W,
    const float* __restrict__ bias, float* __restrict__ out)
{
    __shared__ _Float16 As[2][128][40];
    __shared__ _Float16 Bs[2][128][40];

    const int tid  = threadIdx.x;
    const int wave = tid >> 5;
    const int lane = tid & 31;
    const int l16  = lane & 15;
    const int h8   = lane >> 4;
    const int m0   = blockIdx.y * 128;
    const int n0   = blockIdx.x * 128;

    v8f acc[8];
    for (int t = 0; t < 8; ++t) acc[t] = zero8();

    v4u raA[2];
    float4 rb[4];
    auto gload = [&](int k0) {
        for (int ii = 0; ii < 2; ++ii) {
            const int i = tid + ii * 256;          // 0..511: 128 rows x 4 x16B
            const int r = i >> 2, c8 = (i & 3) << 3;
            raA[ii] = *reinterpret_cast<const v4u*>(
                AO + (size_t)(m0 + r) * DIM + k0 + c8);
        }
        for (int ii = 0; ii < 4; ++ii) {
            const int i = tid + ii * 256;
            const int r = i >> 3, c4 = (i & 7) << 2;
            rb[ii] = *reinterpret_cast<const float4*>(
                W + (size_t)(n0 + r) * DIM + k0 + c4);
        }
    };
    auto lstore = [&](int buf) {
        for (int ii = 0; ii < 2; ++ii) {
            const int i = tid + ii * 256;
            const int r = i >> 2, c8 = (i & 3) << 3;
            *reinterpret_cast<v4u*>(&As[buf][r][c8]) = raA[ii];
        }
        for (int ii = 0; ii < 4; ++ii) {
            const int i = tid + ii * 256;
            const int r = i >> 3, c4 = (i & 7) << 2;
            Bs[buf][r][c4 + 0] = (_Float16)rb[ii].x;
            Bs[buf][r][c4 + 1] = (_Float16)rb[ii].y;
            Bs[buf][r][c4 + 2] = (_Float16)rb[ii].z;
            Bs[buf][r][c4 + 3] = (_Float16)rb[ii].w;
        }
    };
    auto compute = [&](int cur) {
        FragH a;
        const int m = (wave << 4) + l16;
        a.u.lo = *reinterpret_cast<const v4u*>(&As[cur][m][(h8 << 3)]);
        a.u.hi = *reinterpret_cast<const v4u*>(&As[cur][m][16 + (h8 << 3)]);
        FragH bf[8];
        for (int t = 0; t < 8; ++t) {
            const int n = (t << 4) + l16;
            bf[t].u.lo = *reinterpret_cast<const v4u*>(&Bs[cur][n][(h8 << 3)]);
            bf[t].u.hi = *reinterpret_cast<const v4u*>(&Bs[cur][n][16 + (h8 << 3)]);
        }
        SCHED_FENCE();
        for (int t = 0; t < 8; ++t) acc[t] = wmma_f16(a, bf[t], acc[t]);
    };

    gload(0);
    lstore(0);
    __syncthreads();

    for (int kt = 0; kt < DIM / 32 - 1; ++kt) {
        const int cur = kt & 1;
        gload(32 * (kt + 1));
        SCHED_FENCE();
        compute(cur);
        SCHED_FENCE();
        lstore(1 - cur);
        __syncthreads();
    }
    compute((DIM / 32 - 1) & 1);

    for (int t = 0; t < 8; ++t) {
        const int gcol = n0 + (t << 4) + l16;
        const float bt = bias[gcol];
        for (int r = 0; r < 8; ++r) {
            const int grow = m0 + (wave << 4) + r + (h8 << 3);
            out[(size_t)grow * DIM + gcol] = acc[t][r] + bt;
        }
    }
}

// ---------------------------------------------------------------------------
extern "C" void kernel_launch(void* const* d_in, const int* in_sizes, int n_in,
                              void* d_out, int out_size, void* d_ws, size_t ws_size,
                              hipStream_t stream)
{
    const float* H      = (const float*)d_in[0];
    const float* cosp   = (const float*)d_in[1];
    const float* sinp   = (const float*)d_in[2];
    const float* qkv_w  = (const float*)d_in[3];
    const float* qkv_b  = (const float*)d_in[4];
    const float* proj_w = (const float*)d_in[5];
    const float* proj_b = (const float*)d_in[6];
    float* out = (float*)d_out;

    // workspace (f16): Q 8MB | K 8MB | Vt 8MB | AO 8MB = 32MB
    _Float16* Qh = (_Float16*)d_ws;
    _Float16* Kh = Qh + (size_t)NH * SEQ * HD;
    _Float16* Vt = Kh + (size_t)NH * SEQ * HD;
    _Float16* AO = Vt + (size_t)NH * HD * SEQ;

    qkv_rope_kernel<<<dim3(3 * DIM / 128, SEQ / 128), 256, 0, stream>>>(
        H, cosp, sinp, qkv_w, qkv_b, Qh, Kh, Vt);
    attn_kernel<<<dim3(SEQ / 128, NH), 256, 0, stream>>>(Qh, Kh, Vt, AO);
    proj_kernel<<<dim3(DIM / 128, SEQ / 128), 256, 0, stream>>>(
        AO, proj_w, proj_b, out);
}
